// GPTLanguageModel_240518168623
// MI455X (gfx1250) — compile-verified, matrix operand order checked by reading, not run on hardware
//
#include <hip/hip_runtime.h>
#include <hip/hip_bf16.h>

// ---------------------------------------------------------------------------
// Model constants (from reference)
// ---------------------------------------------------------------------------
constexpr int kE  = 1140;          // embed dim
constexpr int kH  = 32;            // heads
constexpr int kHS = 35;            // head size
constexpr int kHD = kH * kHS;      // 1120
constexpr int kL  = 36;            // layers
constexpr int kV  = 45000;         // vocab
constexpr int kT  = 128;           // seq len
constexpr int kB  = 8;             // batch
constexpr int kM  = kB * kT;       // 1024 token rows
constexpr int kFF = 4 * kE;        // 4560
constexpr float kScale = 0.16903085094570331f;  // 35^-0.5

// padded (multiple-of-32) activation row lengths for bf16 buffers
constexpr int kEp  = 1152;   // ceil32(1140)
constexpr int kHDp = 1120;   // already multiple of 32
constexpr int kFFp = 4576;   // ceil32(4560)

// WMMA tiling: block tile 128x64, 4 waves, wave tile 32x64 (acc[2][4])
constexpr int BM  = 128;
constexpr int BN  = 64;
constexpr int BK  = 32;
constexpr int BKP = 40;      // padded LDS row stride (elems): 64B row + 16B pad

typedef __attribute__((ext_vector_type(16))) __bf16 v16bf;
typedef __attribute__((ext_vector_type(8)))  __bf16 v8bf;
typedef __attribute__((ext_vector_type(8)))  float  v8f;
typedef __attribute__((ext_vector_type(4)))  unsigned v4u;
typedef __attribute__((ext_vector_type(8)))  int      v8i;
typedef __attribute__((ext_vector_type(4)))  int      v4i;

// Tensor Data Mover availability (device pass only; host falls back cleanly)
#if defined(__AMDGCN__) && __has_builtin(__builtin_amdgcn_tensor_load_to_lds)
#define USE_TDM 1
#else
#define USE_TDM 0
#endif

// ---------------------------------------------------------------------------
// Embedding gather: x[b*T+t, :] = tok_emb[index[b,t], :] + pos_emb[t, :]
// ---------------------------------------------------------------------------
__global__ __launch_bounds__(256)
void embed_kernel(const int* __restrict__ index,
                  const float* __restrict__ tok_emb,
                  const float* __restrict__ pos_emb,
                  float* __restrict__ x)
{
    const int row = blockIdx.x;
    const int t   = row % kT;
    const int tok = index[row];
    const float* te = tok_emb + (size_t)tok * kE;
    const float* pe = pos_emb + (size_t)t   * kE;
    float* xr = x + (size_t)row * kE;
    for (int e = threadIdx.x; e < kE; e += 256)
        xr[e] = te[e] + pe[e];
}

// ---------------------------------------------------------------------------
// LayerNorm with fused bf16 output (padded, zero-filled rows of length Kp)
// ---------------------------------------------------------------------------
__global__ __launch_bounds__(256)
void layernorm_bf16_kernel(const float* __restrict__ x,
                           const float* __restrict__ g,
                           const float* __restrict__ b,
                           __bf16* __restrict__ ybf, int Kp)
{
    __shared__ float red[256];
    const int row = blockIdx.x;
    const float* xr = x + (size_t)row * kE;
    const int tid = threadIdx.x;

    float s = 0.f;
    for (int e = tid; e < kE; e += 256) s += xr[e];
    red[tid] = s;
    __syncthreads();
    for (int o = 128; o > 0; o >>= 1) {
        if (tid < o) red[tid] += red[tid + o];
        __syncthreads();
    }
    const float mu = red[0] * (1.0f / kE);
    __syncthreads();

    float vs = 0.f;
    for (int e = tid; e < kE; e += 256) {
        float d = xr[e] - mu;
        vs += d * d;
    }
    red[tid] = vs;
    __syncthreads();
    for (int o = 128; o > 0; o >>= 1) {
        if (tid < o) red[tid] += red[tid + o];
        __syncthreads();
    }
    const float rs = rsqrtf(red[0] * (1.0f / kE) + 1e-5f);

    unsigned* dst = (unsigned*)(ybf + (size_t)row * Kp);
    for (int e2 = tid * 2; e2 < Kp; e2 += 512) {
        union { __bf16 h[2]; unsigned u; } pk;
        pk.h[0] = (__bf16)((e2     < kE) ? (xr[e2]     - mu) * rs * g[e2]     + b[e2]     : 0.f);
        pk.h[1] = (__bf16)((e2 + 1 < kE) ? (xr[e2 + 1] - mu) * rs * g[e2 + 1] + b[e2 + 1] : 0.f);
        dst[e2 >> 1] = pk.u;
    }
}

// ---------------------------------------------------------------------------
// bf16 WMMA GEMM:
//   val = A[M,K] * B[K,N] + bias; relu; += resid
//   C   (fp32, optional)  <- val
//   Cbf (bf16, optional, padded width Np, zero-filled pad) <- val
//  - A: bf16, rows padded to lda (multiple of 32, zero-filled) -> no guards;
//    A tile staged into LDS by the Tensor Data Mover (pad_amount produces the
//    80B padded LDS rows); falls back to per-thread b128 copies without TDM.
//  - B: fp32 weights, converted to bf16 during LDS staging (per-thread).
//  - LDS double-buffered; uniform interior/tail path selection.
// Block: 128 threads (4 waves). Wave tile 32x64: 8 WMMAs / 12 ds_load_b128.
// ---------------------------------------------------------------------------
__global__ __launch_bounds__(128)
void gemm_bf16_kernel(const __bf16* __restrict__ A, int lda,
                      const float* __restrict__ Bm,
                      const float* __restrict__ bias,    // nullptr ok (len N)
                      const float* __restrict__ resid,   // nullptr ok [M,N]
                      float* __restrict__ C,             // nullptr ok [M,N]
                      __bf16* __restrict__ Cbf,          // nullptr ok [M,Np]
                      int Np,
                      int M, int N, int K, int relu_flag)
{
    __shared__ __bf16 As[2][BM][BKP];     // 2 x 10 KB
    __shared__ __bf16 Bs[2][BN][BKP];     // 2 x 5 KB (transposed [n][k])

    const int tid  = threadIdx.x;
    const int wave = tid >> 5;
    const int lane = tid & 31;
    const int bm   = blockIdx.y * BM;
    const int bn   = blockIdx.x * BN;

    const int kb = lane >> 4;   // K-half select (16-bit A/B operand layout)
    const int ml = lane & 15;   // row (A) / col (B) within 16-tile

    const bool nfull = (bn + BN) <= N;

#if USE_TDM
    const unsigned ldsA[2] = { (unsigned)(uintptr_t)&As[0][0][0],
                               (unsigned)(uintptr_t)&As[1][0][0] };
    const int d0 = lda >> 1;                 // tensor row length/stride (DWORDs)
#endif

    auto stage = [&](int p, int k0) {
#if USE_TDM
        // --- A tile via TDM: 2D tile, 16 DW x 128 rows, LDS rows padded +4 DW
        if (wave == 0) {
            const unsigned long long ga =
                (unsigned long long)(uintptr_t)(A + (size_t)bm * lda + k0);
            v4u g0 = { 1u,                                  // count=1 D#
                       ldsA[p],                             // lds_addr
                       (unsigned)ga,                        // global_addr[31:0]
                       ((unsigned)(ga >> 32) & 0x01FFFFFFu) // global_addr[56:32]
                           | 0x80000000u };                 // type=2 (image)
            v8i g1 = { (int)0x06D20000,        // data_size=4B, pad_en,
                                               // pad_interval=16DW, pad_amt=4DW
                       (int)((d0 & 0xFFFF) << 16),  // tensor_dim0 (low16)
                       (int)(128 << 16),            // tensor_dim1 = 128 rows
                       (int)(16  << 16),            // tile_dim0 = 16 DWORDs
                       (int)128,                    // tile_dim1 = 128 rows
                       (int)d0,                     // tensor_dim0_stride (DWORDs)
                       0, 0 };
            v4i z4 = { 0, 0, 0, 0 };
#if __clang_major__ >= 23
            v8i z8 = { 0, 0, 0, 0, 0, 0, 0, 0 };
            __builtin_amdgcn_tensor_load_to_lds(g0, g1, z4, z4, z8, 0);
#else
            __builtin_amdgcn_tensor_load_to_lds(g0, g1, z4, z4, 0);
#endif
        }
#else
        {   // A tile: 128 rows x 32 k of bf16; one full row per thread (64B)
            const v8bf* src = (const v8bf*)(A + (size_t)(bm + tid) * lda + k0);
            const v8bf r0 = src[0], r1 = src[1], r2 = src[2], r3 = src[3];
            *(v8bf*)&As[p][tid][0]  = r0;
            *(v8bf*)&As[p][tid][8]  = r1;
            *(v8bf*)&As[p][tid][16] = r2;
            *(v8bf*)&As[p][tid][24] = r3;
        }
#endif
        {   // B tile: thread gathers one column (n) over 16 k, stores 32B run
            const int nI = tid & 63;
            const int kh = tid >> 6;              // 0 or 1
            const int kbase = k0 + kh * 16;
            alignas(16) __bf16 col[16];
            if (nfull && (k0 + BK) <= K) {        // uniform fast path
                const float* src = Bm + (size_t)kbase * N + bn + nI;
#pragma unroll
                for (int i = 0; i < 16; ++i)
                    col[i] = (__bf16)src[(size_t)i * N];
            } else {                              // uniform tail path
                const int n = bn + nI;
#pragma unroll
                for (int i = 0; i < 16; ++i) {
                    const int kk = kbase + i;
                    col[i] = (kk < K && n < N)
                               ? (__bf16)Bm[(size_t)kk * N + n] : (__bf16)0.f;
                }
            }
            *(v8bf*)&Bs[p][nI][kh * 16]     = *(const v8bf*)&col[0];
            *(v8bf*)&Bs[p][nI][kh * 16 + 8] = *(const v8bf*)&col[8];
        }
    };

    v8f acc[2][4];
#pragma unroll
    for (int r = 0; r < 2; ++r)
#pragma unroll
        for (int t = 0; t < 4; ++t)
#pragma unroll
            for (int j = 0; j < 8; ++j) acc[r][t][j] = 0.f;

    stage(0, 0);
#if USE_TDM
    __builtin_amdgcn_s_wait_tensorcnt(0);
#endif
    __syncthreads();

    int p = 0;
    for (int k0 = 0; k0 < K; k0 += BK) {
        if (k0 + BK < K) {
            __builtin_prefetch((const void*)(Bm + (size_t)(k0 + BK) * N + bn), 0, 3);
            stage(p ^ 1, k0 + BK);
        }

        // A fragments: lanes 0-15 hold K {0..7,16..23}; 16-31 hold {8..15,24..31}
        v16bf afrag[2];
#pragma unroll
        for (int r = 0; r < 2; ++r) {
            const __bf16* arow = &As[p][wave * 32 + r * 16 + ml][0];
            const v8bf alo = *(const v8bf*)(arow + kb * 8);
            const v8bf ahi = *(const v8bf*)(arow + 16 + kb * 8);
#pragma unroll
            for (int i = 0; i < 8; ++i) {
                afrag[r][i] = alo[i]; afrag[r][8 + i] = ahi[i];
            }
        }

#pragma unroll
        for (int t = 0; t < 4; ++t) {
            const __bf16* brow = &Bs[p][t * 16 + ml][0];
            const v8bf blo = *(const v8bf*)(brow + kb * 8);
            const v8bf bhi = *(const v8bf*)(brow + 16 + kb * 8);
            v16bf bfrag;
#pragma unroll
            for (int i = 0; i < 8; ++i) { bfrag[i] = blo[i]; bfrag[8 + i] = bhi[i]; }

            acc[0][t] = __builtin_amdgcn_wmma_f32_16x16x32_bf16(
                false, afrag[0], false, bfrag, (short)0, acc[0][t], false, false);
            acc[1][t] = __builtin_amdgcn_wmma_f32_16x16x32_bf16(
                false, afrag[1], false, bfrag, (short)0, acc[1][t], false, false);
        }
#if USE_TDM
        __builtin_amdgcn_s_wait_tensorcnt(0);
#endif
        __syncthreads();
        p ^= 1;
    }

    // epilogue (C/D layout: VGPR v holds M = v + 8*(lane>>4), N = lane&15)
    const int nl = lane & 15;
#pragma unroll
    for (int r = 0; r < 2; ++r) {
        const int mbase = bm + wave * 32 + r * 16 + (lane >> 4) * 8;
#pragma unroll
        for (int t = 0; t < 4; ++t) {
            const int n = bn + t * 16 + nl;
            const bool nin = n < N;
            const float bv = (bias && nin) ? bias[n] : 0.f;
#pragma unroll
            for (int v = 0; v < 8; ++v) {
                const int m = mbase + v;
                float val = acc[r][t][v] + bv;
                if (relu_flag) val = fmaxf(val, 0.f);
                if (resid && nin) val += resid[(size_t)m * N + n];
                if (C && nin) C[(size_t)m * N + n] = val;
                if (Cbf && n < Np)
                    Cbf[(size_t)m * Np + n] = (__bf16)(nin ? val : 0.f);
            }
        }
    }
}

// ---------------------------------------------------------------------------
// Causal attention for one (b,h): K/V staged into LDS with CDNA5 async
// global->LDS copies (ASYNCcnt), then per-row online softmax. bf16 output.
// ---------------------------------------------------------------------------
__global__ __launch_bounds__(128)
void attention_kernel(const float* __restrict__ q,
                      const float* __restrict__ k,
                      const float* __restrict__ v,
                      __bf16* __restrict__ att)
{
    __shared__ float ks[kT][kHS + 1];
    __shared__ float vs[kT][kHS + 1];

    const int b = blockIdx.x / kH;
    const int h = blockIdx.x % kH;
    const size_t base = (size_t)b * kT * kHD + (size_t)h * kHS;
    const int tid = threadIdx.x;

    // async global->LDS staging (4480 elements each, 35 full rounds of 128)
    for (int i = tid; i < kT * kHS; i += 128) {
        const int r = i / kHS, d = i % kHS;
        const float* gk = k + base + (size_t)r * kHD + d;
        const float* gv = v + base + (size_t)r * kHD + d;
        const unsigned lk = (unsigned)(uintptr_t)&ks[r][d];
        const unsigned lv = (unsigned)(uintptr_t)&vs[r][d];
        asm volatile("global_load_async_to_lds_b32 %0, %1, off"
                     :: "v"(lk), "v"(gk) : "memory");
        asm volatile("global_load_async_to_lds_b32 %0, %1, off"
                     :: "v"(lv), "v"(gv) : "memory");
    }
    asm volatile("s_wait_asynccnt 0x0" ::: "memory");
    __syncthreads();

    float qr[kHS];
#pragma unroll
    for (int d = 0; d < kHS; ++d) qr[d] = q[base + (size_t)tid * kHD + d];

    float mrun = -1e30f, lrun = 0.f;
    float accv[kHS];
#pragma unroll
    for (int d = 0; d < kHS; ++d) accv[d] = 0.f;

    for (int s = 0; s <= tid; ++s) {
        float z = 0.f;
#pragma unroll
        for (int d = 0; d < kHS; ++d) z += qr[d] * ks[s][d];
        z *= kScale;
        const float mn   = fmaxf(mrun, z);
        const float corr = __expf(mrun - mn);
        const float pw   = __expf(z - mn);
        lrun = lrun * corr + pw;
#pragma unroll
        for (int d = 0; d < kHS; ++d) accv[d] = accv[d] * corr + pw * vs[s][d];
        mrun = mn;
    }

    const float inv = 1.f / lrun;
#pragma unroll
    for (int d = 0; d < kHS; ++d)
        att[base + (size_t)tid * kHD + d] = (__bf16)(accv[d] * inv);
}

// ---------------------------------------------------------------------------
// Host-side orchestration
// ---------------------------------------------------------------------------
static inline dim3 gemm_grid(int N) { return dim3((N + BN - 1) / BN, kM / BM); }

extern "C" void kernel_launch(void* const* d_in, const int* in_sizes, int n_in,
                              void* d_out, int out_size, void* d_ws, size_t ws_size,
                              hipStream_t stream)
{
    (void)in_sizes; (void)n_in; (void)out_size; (void)ws_size;

    const int*   index   = (const int*)  d_in[0];
    const float* tok_emb = (const float*)d_in[1];
    const float* pos_emb = (const float*)d_in[2];
    const float* wq      = (const float*)d_in[3];
    const float* wk      = (const float*)d_in[4];
    const float* wv      = (const float*)d_in[5];
    const float* wo      = (const float*)d_in[6];
    const float* bo      = (const float*)d_in[7];
    const float* ln1g    = (const float*)d_in[8];
    const float* ln1b    = (const float*)d_in[9];
    const float* ln2g    = (const float*)d_in[10];
    const float* ln2b    = (const float*)d_in[11];
    const float* w1      = (const float*)d_in[12];
    const float* b1      = (const float*)d_in[13];
    const float* w2      = (const float*)d_in[14];
    const float* b2      = (const float*)d_in[15];
    const float* lnfg    = (const float*)d_in[16];
    const float* lnfb    = (const float*)d_in[17];
    const float* wh      = (const float*)d_in[18];
    const float* bh      = (const float*)d_in[19];
    float* out = (float*)d_out;

    // workspace carve-out (256B aligned)
    char* ws = (char*)d_ws;
    size_t off = 0;
    auto take = [&](size_t bytes) -> char* {
        char* p = ws + off;
        off += (bytes + 255) & ~(size_t)255;
        return p;
    };
    float*  x     = (float*) take((size_t)kM * kE   * 4);
    float*  qb    = (float*) take((size_t)kM * kHD  * 4);
    float*  kbuf  = (float*) take((size_t)kM * kHD  * 4);
    float*  vbuf  = (float*) take((size_t)kM * kHD  * 4);
    __bf16* ybf   = (__bf16*)take((size_t)kM * kEp  * 2);
    __bf16* attbf = (__bf16*)take((size_t)kM * kHDp * 2);
    __bf16* hbf   = (__bf16*)take((size_t)kM * kFFp * 2);

    embed_kernel<<<kM, 256, 0, stream>>>(index, tok_emb, pos_emb, x);

    for (int l = 0; l < kL; ++l) {
        const float* lwq = wq + (size_t)l * kE  * kHD;
        const float* lwk = wk + (size_t)l * kE  * kHD;
        const float* lwv = wv + (size_t)l * kE  * kHD;
        const float* lwo = wo + (size_t)l * kHD * kE;
        const float* lbo = bo + (size_t)l * kE;
        const float* lw1 = w1 + (size_t)l * kE  * kFF;
        const float* lb1 = b1 + (size_t)l * kFF;
        const float* lw2 = w2 + (size_t)l * kFF * kE;
        const float* lb2 = b2 + (size_t)l * kE;

        // ybf = bf16(LN1(x))  (feeds 3 GEMMs)
        layernorm_bf16_kernel<<<kM, 256, 0, stream>>>(
            x, ln1g + (size_t)l * kE, ln1b + (size_t)l * kE, ybf, kEp);

        gemm_bf16_kernel<<<gemm_grid(kHD), 128, 0, stream>>>(
            ybf, kEp, lwq, nullptr, nullptr, qb, nullptr, 0, kM, kHD, kE, 0);
        gemm_bf16_kernel<<<gemm_grid(kHD), 128, 0, stream>>>(
            ybf, kEp, lwk, nullptr, nullptr, kbuf, nullptr, 0, kM, kHD, kE, 0);
        gemm_bf16_kernel<<<gemm_grid(kHD), 128, 0, stream>>>(
            ybf, kEp, lwv, nullptr, nullptr, vbuf, nullptr, 0, kM, kHD, kE, 0);

        // attbf = bf16(causal softmax(q k^T * scale) v)
        attention_kernel<<<kB * kH, 128, 0, stream>>>(qb, kbuf, vbuf, attbf);

        // x = x + att @ wo + bo
        gemm_bf16_kernel<<<gemm_grid(kE), 128, 0, stream>>>(
            attbf, kHDp, lwo, lbo, x, x, nullptr, 0, kM, kE, kHD, 0);

        // ybf = bf16(LN2(x)); hbf = bf16(relu(y @ w1 + b1)); x = x + h @ w2 + b2
        layernorm_bf16_kernel<<<kM, 256, 0, stream>>>(
            x, ln2g + (size_t)l * kE, ln2b + (size_t)l * kE, ybf, kEp);
        gemm_bf16_kernel<<<gemm_grid(kFF), 128, 0, stream>>>(
            ybf, kEp, lw1, lb1, nullptr, nullptr, hbf, kFFp, kM, kFF, kE, 1);
        gemm_bf16_kernel<<<gemm_grid(kE), 128, 0, stream>>>(
            hbf, kFFp, lw2, lb2, x, x, nullptr, 0, kM, kE, kFF, 0);
    }

    // final LN + vocab head
    layernorm_bf16_kernel<<<kM, 256, 0, stream>>>(x, lnfg, lnfb, ybf, kEp);
    gemm_bf16_kernel<<<gemm_grid(kV), 128, 0, stream>>>(
        ybf, kEp, wh, bh, nullptr, out, nullptr, 0, kM, kV, kE, 0);
}
